// AdaptiveForgettingController_55087250538929
// MI455X (gfx1250) — matrix operand order repaired and checked by reading, not compiled, not run on hardware
//
#include <hip/hip_runtime.h>

#define S_MEM 65536
#define D_DIM 512
#define B_DIM 64

typedef __attribute__((ext_vector_type(16))) __bf16 v16bf;
typedef __attribute__((ext_vector_type(8)))  float  v8f;
typedef int v4i __attribute__((vector_size(16)));   // matches async-LDS builtin param

union BF16x16 { v16bf v; __bf16 e[16]; uint4 u4[2]; };

__device__ inline float sigmoidf(float x) { return 1.f / (1.f + __expf(-x)); }

// Async global->LDS path (gfx1250) if the toolchain declares the builtin.
#if defined(__HIP_DEVICE_COMPILE__) && __has_builtin(__builtin_amdgcn_global_load_async_to_lds_b128)
#define HAVE_ASYNC_LDS 1
#else
#define HAVE_ASYNC_LDS 0
#endif

// ---- scalar workspace slots (float indices; u64 packs 8-byte aligned) ----
#define SC_ACTIVE   0   // int: count(norms > 0.5)
#define SC_AGEMAX   1   // uint: bits of max age
#define SC_VPACK    2   // u64 @ [2,3]: argmax es_em
#define SC_NOVSUM   4
#define SC_SCORESUM 5
#define SC_CONDCNT  6   // int
#define SC_MPACK    8   // u64: argmax masked
#define SC_APACK    10  // u64: argmax slot_age
#define SC_CHSUM    12
#define SC_NOVMEAN  16
#define SC_CAP      17
#define SC_DYNTHR   18
#define SC_TOPKTHR  19
#define SC_RAWTHR   20
#define SC_DOERASE  21  // int
#define SC_VICTIM   22  // int
#define SC_NOVOK    23  // int
#define SC_BASEOK   24  // int
#define SC_SHOULD   25  // int
#define SC_WIDX     26  // int
#define SC_USEMASK  27  // int
#define SC_USESLOT  28  // int
#define SC_NACTIVE  29  // int

__global__ void k_init(float* scal) {
  if (threadIdx.x < 64) scal[threadIdx.x] = 0.f;
}

// fp32 -> bf16 convert (weights)
__global__ __launch_bounds__(256) void k_cvt(const float* __restrict__ src,
                                             __bf16* __restrict__ dst, int n) {
  for (int i = blockIdx.x * 256 + threadIdx.x; i < n; i += gridDim.x * 256)
    dst[i] = (__bf16)src[i];
}

// per-b prep: normalized new_content (bf16), raw new_content (bf16), combined=[nc,q] (bf16)
__global__ __launch_bounds__(128) void k_prep(const float* __restrict__ nc,
                                              const float* __restrict__ q,
                                              __bf16* __restrict__ nn_bf,
                                              __bf16* __restrict__ nc_bf,
                                              __bf16* __restrict__ comb_bf) {
  __shared__ float red[128];
  int b = blockIdx.x, t = threadIdx.x;
  const float* row = nc + (size_t)b * D_DIM;
  float s = 0.f;
  for (int i = t; i < D_DIM; i += 128) { float x = row[i]; s += x * x; }
  red[t] = s; __syncthreads();
  for (int o = 64; o > 0; o >>= 1) { if (t < o) red[t] += red[t + o]; __syncthreads(); }
  float inv = 1.f / fmaxf(sqrtf(red[0]), 1e-12f);
  for (int i = t; i < D_DIM; i += 128) {
    float x = row[i];
    nn_bf[(size_t)b * D_DIM + i] = (__bf16)(x * inv);
    nc_bf[(size_t)b * D_DIM + i] = (__bf16)x;
    comb_bf[(size_t)b * 1024 + i] = (__bf16)x;
    comb_bf[(size_t)b * 1024 + 512 + i] = (__bf16)q[(size_t)b * D_DIM + i];
  }
}

// ---- fused heavy pass: one read of memory_bank tile -> norms + sims + conf ----
// Tile is staged (async DMA to LDS if available) then converted ONCE to a bf16 LDS
// tile so the WMMA inner loops are pure ds_load_b128 + v_wmma (no per-fragment cvt).
__global__ __launch_bounds__(128) void k_heavy(
    const float* __restrict__ mem, const __bf16* __restrict__ nn_bf,
    const __bf16* __restrict__ ecw_bf, const float* __restrict__ ecb,
    float* __restrict__ sims, float* __restrict__ norms,
    float* __restrict__ conf32, float* __restrict__ scal) {
  __shared__ __bf16 tileB[16][520];  // bf16 tile for WMMA fragments
  __shared__ float norm2[16];
  __shared__ float inv[16];
  __shared__ int cnt;
#if HAVE_ASYNC_LDS
  __shared__ float tileF[16][516];   // fp32 landing zone for async copy
#endif
  const int t = threadIdx.x;
  const int s0 = blockIdx.x << 4;
  if (t < 16) norm2[t] = 0.f;
  if (t == 0) cnt = 0;
  if (t < 64) __builtin_prefetch(nn_bf + (size_t)t * D_DIM, 0, 2);  // global_prefetch_b8
  __syncthreads();

  // 8 threads per row; each thread owns 64 contiguous columns of one row
  const int r = t >> 3, seg = t & 7;
  const float* grow = mem + (size_t)(s0 + r) * D_DIM + seg * 64;
  float sq = 0.f;
#if HAVE_ASYNC_LDS
  // async DMA of the fp32 tile straight into LDS (ASYNCcnt-tracked)
  #pragma unroll
  for (int j = 0; j < 16; ++j) {
    __builtin_amdgcn_global_load_async_to_lds_b128(
        (__attribute__((address_space(1))) v4i*)(grow + j * 4),
        (__attribute__((address_space(3))) v4i*)&tileF[r][seg * 64 + j * 4],
        0, 0);
  }
#if __has_builtin(__builtin_amdgcn_s_wait_asynccnt)
  __builtin_amdgcn_s_wait_asynccnt(0);
#else
  asm volatile("s_wait_asynccnt 0x0" ::: "memory");
#endif
  __syncthreads();
  // LDS fp32 -> LDS bf16 (single conversion) + row sum-of-squares
  #pragma unroll
  for (int j = 0; j < 16; ++j) {
    float4 v = *(const float4*)&tileF[r][seg * 64 + j * 4];
    sq += v.x * v.x + v.y * v.y + v.z * v.z + v.w * v.w;
    union { __bf16 e[4]; uint2 u; } c;
    c.e[0] = (__bf16)v.x; c.e[1] = (__bf16)v.y;
    c.e[2] = (__bf16)v.z; c.e[3] = (__bf16)v.w;
    *(uint2*)&tileB[r][seg * 64 + j * 4] = c.u;
  }
#else
  // fallback: global -> VGPR -> bf16 LDS, sumsq from registers
  #pragma unroll
  for (int j = 0; j < 16; ++j) {
    float4 v = *(const float4*)(grow + j * 4);
    sq += v.x * v.x + v.y * v.y + v.z * v.z + v.w * v.w;
    union { __bf16 e[4]; uint2 u; } c;
    c.e[0] = (__bf16)v.x; c.e[1] = (__bf16)v.y;
    c.e[2] = (__bf16)v.z; c.e[3] = (__bf16)v.w;
    *(uint2*)&tileB[r][seg * 64 + j * 4] = c.u;
  }
#endif
  atomicAdd(&norm2[r], sq);
  __syncthreads();
  if (t < 16) {
    float n = sqrtf(norm2[t]);
    norms[s0 + t] = n;
    inv[t] = 1.f / fmaxf(n, 1e-12f);
    if (n > 0.5f) atomicAdd(&cnt, 1);
  }
  __syncthreads();
  if (t == 0 && cnt) atomicAdd((int*)scal + SC_ACTIVE, cnt);

  const int lane = t & 31, wave = t >> 5;
  const int hi = lane >> 4, l15 = lane & 15;
  // sims: each wave owns a 16-row M tile of nn; B = memT tile from bf16 LDS
  {
    v8f acc = {};
    const __bf16* arow = nn_bf + (size_t)((wave << 4) + l15) * D_DIM;
    const __bf16* brow = tileB[l15];
    for (int kk = 0; kk < D_DIM; kk += 32) {
      BF16x16 a, b;
      int ka = kk + hi * 8;
      a.u4[0] = *(const uint4*)(arow + ka);
      a.u4[1] = *(const uint4*)(arow + ka + 16);
      const __bf16* bp = brow + kk + hi * 8;
      b.u4[0] = *(const uint4*)(bp);
      b.u4[1] = *(const uint4*)(bp + 16);
      acc = __builtin_amdgcn_wmma_f32_16x16x32_bf16(false, a.v, false, b.v,
                                                    (short)0, acc, false, false);
    }
    float sc = inv[l15];
    float* crow = sims + (size_t)((wave << 4) + hi * 8) * S_MEM + s0 + l15;
    #pragma unroll
    for (int rr = 0; rr < 8; ++rr) crow[(size_t)rr * S_MEM] = acc[rr] * sc;
  }
  // conf: waves 0,1 each own a 16-col N tile of ec_w; A = mem tile from bf16 LDS
  if (wave < 2) {
    v8f acc = {};
    const __bf16* arow2 = tileB[l15];
    for (int kk = 0; kk < D_DIM; kk += 32) {
      BF16x16 a, b;
      const __bf16* ap = arow2 + kk + hi * 8;
      a.u4[0] = *(const uint4*)(ap);
      a.u4[1] = *(const uint4*)(ap + 16);
      const __bf16* bp = ecw_bf + (size_t)(kk + hi * 8) * 32 + (wave << 4) + l15;
      #pragma unroll
      for (int rr = 0; rr < 16; ++rr) b.e[rr] = bp[(size_t)(rr + (rr >= 8 ? 8 : 0)) * 32];
      acc = __builtin_amdgcn_wmma_f32_16x16x32_bf16(false, a.v, false, b.v,
                                                    (short)0, acc, false, false);
    }
    int j = (wave << 4) + l15;
    float bj = ecb[j];
    float* crow = conf32 + (size_t)(s0 + hi * 8) * 32 + j;
    #pragma unroll
    for (int rr = 0; rr < 8; ++rr) crow[(size_t)rr * 32] = sigmoidf(acc[rr] + bj);
  }
}

// ---- generic bf16 WMMA GEMM: C(MxN) = A(MxK, row-major) @ B(KxN, row-major) ----
__global__ __launch_bounds__(128) void k_wmma_gemm(
    const __bf16* __restrict__ A, const __bf16* __restrict__ B,
    float* __restrict__ C, int M, int N, int K) {
  const int lane = threadIdx.x & 31, wave = threadIdx.x >> 5;
  const int ntiles = N >> 4;
  const int tile = blockIdx.x * 4 + wave;
  if (tile >= (M >> 4) * ntiles) return;
  const int m0 = (tile / ntiles) << 4;
  const int n0 = (tile % ntiles) << 4;
  const int hi = lane >> 4, l15 = lane & 15;
  v8f acc = {};
  const __bf16* arow = A + (size_t)(m0 + l15) * K;
  for (int kk = 0; kk < K; kk += 32) {
    BF16x16 a, b;
    int ka = kk + hi * 8;
    a.u4[0] = *(const uint4*)(arow + ka);
    a.u4[1] = *(const uint4*)(arow + ka + 16);
    const __bf16* bp = B + (size_t)(kk + hi * 8) * N + n0 + l15;
    #pragma unroll
    for (int rr = 0; rr < 16; ++rr) b.e[rr] = bp[(size_t)(rr + (rr >= 8 ? 8 : 0)) * N];
    acc = __builtin_amdgcn_wmma_f32_16x16x32_bf16(false, a.v, false, b.v,
                                                  (short)0, acc, false, false);
  }
  float* crow = C + (size_t)(m0 + hi * 8) * N + n0 + l15;
  #pragma unroll
  for (int rr = 0; rr < 8; ++rr) crow[(size_t)rr * N] = acc[rr];
}

// layernorm(h+b1)*g+beta -> relu -> bf16
__global__ __launch_bounds__(256) void k_b_ln(const float* __restrict__ Hpre,
    const float* __restrict__ b1, const float* __restrict__ g,
    const float* __restrict__ beta, __bf16* __restrict__ hrelu) {
  __shared__ float red[256];
  int b = blockIdx.x, t = threadIdx.x;
  float x = Hpre[(size_t)b * 256 + t] + b1[t];
  red[t] = x; __syncthreads();
  for (int o = 128; o > 0; o >>= 1) { if (t < o) red[t] += red[t + o]; __syncthreads(); }
  float m = red[0] / 256.f; __syncthreads();
  float d = x - m;
  red[t] = d * d; __syncthreads();
  for (int o = 128; o > 0; o >>= 1) { if (t < o) red[t] += red[t + o]; __syncthreads(); }
  float v = red[0] / 256.f;
  float y = d * rsqrtf(v + 1e-5f) * g[t] + beta[t];
  hrelu[(size_t)b * 256 + t] = (__bf16)fmaxf(y, 0.f);
}

__global__ void k_b_score(const float* __restrict__ relpre, const float* __restrict__ b2,
    const float* __restrict__ nfpre, const float* __restrict__ snb,
    const float* __restrict__ sgw, const float* __restrict__ sgb,
    float* __restrict__ out_score, float* __restrict__ scal) {
  int b = threadIdx.x;
  if (b >= B_DIM) return;
  float acc = sgb[0];
  for (int j = 0; j < 128; ++j) {
    float rel = fmaxf(relpre[b * 128 + j] + b2[j], 0.f);
    float nf  = sigmoidf(nfpre[b * 128 + j] + snb[j]);
    acc += (rel + nf) * sgw[j];
  }
  float v = sigmoidf(acc);
  out_score[b] = v;
  atomicAdd(&scal[SC_SCORESUM], v);
}

__global__ __launch_bounds__(256) void k_age(const float* __restrict__ at,
                                             const int* __restrict__ stepp, float* scal) {
  __shared__ float red[256];
  int t = threadIdx.x, s = blockIdx.x * 256 + t;
  float step = (float)stepp[0];
  red[t] = fmaxf(step - at[s], 0.f);
  __syncthreads();
  for (int o = 128; o > 0; o >>= 1) { if (t < o) red[t] = fmaxf(red[t], red[t + o]); __syncthreads(); }
  if (t == 0) atomicMax((unsigned*)&scal[SC_AGEMAX], __float_as_uint(red[0]));
}

__global__ __launch_bounds__(256) void k_victim(const float* __restrict__ at,
    const float* __restrict__ norms, const int* __restrict__ stepp, float* scal) {
  __shared__ unsigned long long red[256];
  int t = threadIdx.x, s = blockIdx.x * 256 + t;
  float step = (float)stepp[0];
  float agemax = __uint_as_float(*(const unsigned*)&scal[SC_AGEMAX]);
  float age = fmaxf(step - at[s], 0.f);
  float es = age / (agemax + 1e-6f) + (1.f - sigmoidf(norms[s]));
  red[t] = ((unsigned long long)__float_as_uint(es) << 32) | (unsigned)(0xFFFFFFFFu - (unsigned)s);
  __syncthreads();
  for (int o = 128; o > 0; o >>= 1) {
    if (t < o) red[t] = red[t] > red[t + o] ? red[t] : red[t + o];
    __syncthreads();
  }
  if (t == 0) atomicMax((unsigned long long*)&scal[SC_VPACK], red[0]);
}

__global__ __launch_bounds__(256) void k_novelty(const float* __restrict__ sims,
                                                 float* __restrict__ out_nov, float* scal) {
  __shared__ float red[256];
  int b = blockIdx.x, t = threadIdx.x;
  const float* row = sims + (size_t)b * S_MEM;
  float m = -1e30f;
  for (int s = t; s < S_MEM; s += 256) m = fmaxf(m, row[s]);
  red[t] = m; __syncthreads();
  for (int o = 128; o > 0; o >>= 1) { if (t < o) red[t] = fmaxf(red[t], red[t + o]); __syncthreads(); }
  if (t == 0) {
    float nov = (1.f - red[0]) * 0.5f;
    out_nov[b] = nov;
    atomicAdd(&scal[SC_NOVSUM], nov);
  }
}

__global__ void k_decide1(float* scal) {
  int* si = (int*)scal;
  unsigned long long vpack = *(unsigned long long*)&scal[SC_VPACK];
  int n_active = si[SC_ACTIVE];
  float cap = (float)n_active / (float)S_MEM;
  float dyn = cap < 0.3f ? 0.08f : (cap < 0.6f ? 0.08f + (cap - 0.3f) * 0.733f
                                               : 0.3f + (cap - 0.6f));
  dyn = fminf(fmaxf(dyn, 0.f), 0.7f);
  float rawthr = cap < 0.3f ? 0.3f : 0.5f;
  float novmean = scal[SC_NOVSUM] / (float)B_DIM;
  scal[SC_DYNTHR] = dyn;
  scal[SC_TOPKTHR] = cap < 0.3f ? 0.1f : (cap < 0.6f ? 0.2f : 0.4f);
  scal[SC_RAWTHR] = rawthr;
  scal[SC_NOVMEAN] = novmean;
  scal[SC_CAP] = cap;
  si[SC_NACTIVE] = n_active;
  si[SC_DOERASE] = cap > 0.85f;
  si[SC_VICTIM]  = (int)(0xFFFFFFFFu - (unsigned)(vpack & 0xFFFFFFFFull));
  si[SC_NOVOK]   = novmean > dyn;
  si[SC_BASEOK]  = (scal[SC_SCORESUM] / (float)B_DIM) > rawthr;
}

__global__ __launch_bounds__(256) void k_cond(const float* __restrict__ sims,
                                              const float* __restrict__ norms, float* scal) {
  __shared__ int red[256];
  float novmean = scal[SC_NOVMEAN];
  int t = threadIdx.x, cnt = 0;
  for (int i = blockIdx.x * 256 + t; i < B_DIM * S_MEM; i += gridDim.x * 256) {
    int s = i & (S_MEM - 1);
    if ((novmean > 1.f - sims[i]) && (norms[s] > 0.5f)) cnt++;
  }
  red[t] = cnt; __syncthreads();
  for (int o = 128; o > 0; o >>= 1) { if (t < o) red[t] += red[t + o]; __syncthreads(); }
  if (t == 0 && red[0]) atomicAdd((int*)scal + SC_CONDCNT, red[0]);
}

__global__ void k_decide2(float* scal) {
  int* si = (int*)scal;
  int n_active = si[SC_NACTIVE];
  float perc;
  if (n_active > 0) {
    float denom = (float)B_DIM * (float)(n_active > 1 ? n_active : 1);
    perc = (float)si[SC_CONDCNT] / denom;
  } else perc = 1.f;
  int topk_ok = perc > scal[SC_TOPKTHR];
  int should = si[SC_BASEOK] && si[SC_NOVOK] && topk_ok;
  if (si[SC_DOERASE] && !si[SC_NOVOK]) should = 0;
  si[SC_SHOULD] = should;
}

__global__ void k_erase_apply(float* out_mem, float* out_at, const float* scal) {
  const int* si = (const int*)scal;
  if (!si[SC_DOERASE]) return;
  int v = si[SC_VICTIM];
  out_mem[(size_t)v * D_DIM + threadIdx.x] = 0.f;
  if (threadIdx.x == 0) out_at[v] = -99999.f;
}

__global__ __launch_bounds__(256) void k_es(const float* __restrict__ at,
    const float* __restrict__ conf32, const float* __restrict__ elw,
    const float* __restrict__ elb, const float* __restrict__ egw,
    const float* __restrict__ egb, const float* __restrict__ ecb,
    const float* __restrict__ scal, const int* __restrict__ stepp,
    float* __restrict__ es) {
  int s = blockIdx.x * 256 + threadIdx.x;
  if (s >= S_MEM) return;
  const int* si = (const int*)scal;
  float step = (float)stepp[0];
  float x = (step - at[s]) / 1000.f;
  bool vic = si[SC_DOERASE] && (s == si[SC_VICTIM]);
  float acc = egb[0];
  for (int j = 0; j < 32; ++j) {
    float lru = fmaxf(x * elw[j] + elb[j], 0.f);
    float cj = vic ? sigmoidf(ecb[j]) : conf32[(size_t)s * 32 + j];
    acc += lru * egw[j] + cj * egw[32 + j];
  }
  es[s] = sigmoidf(acc);
}

__global__ __launch_bounds__(256) void k_topk(const float* __restrict__ sims,
    const float* __restrict__ scal, int* __restrict__ ti, float* __restrict__ tv) {
  __shared__ float sv[768];
  __shared__ int   si_[768];
  int b = blockIdx.x, t = threadIdx.x;
  const int* si = (const int*)scal;
  int doe = si[SC_DOERASE], vic = si[SC_VICTIM];
  float v0 = -1e30f, v1 = -1e30f, v2 = -1e30f; int i0 = 0, i1 = 0, i2 = 0;
  const float* row = sims + (size_t)b * S_MEM;
  for (int s = t; s < S_MEM; s += 256) {
    float v = row[s];
    if (doe && s == vic) v = 0.f;
    if (v > v0)      { v2 = v1; i2 = i1; v1 = v0; i1 = i0; v0 = v; i0 = s; }
    else if (v > v1) { v2 = v1; i2 = i1; v1 = v; i1 = s; }
    else if (v > v2) { v2 = v; i2 = s; }
  }
  sv[t * 3] = v0; si_[t * 3] = i0;
  sv[t * 3 + 1] = v1; si_[t * 3 + 1] = i1;
  sv[t * 3 + 2] = v2; si_[t * 3 + 2] = i2;
  __syncthreads();
  if (t == 0) {
    float w0 = -1e30f, w1 = -1e30f, w2 = -1e30f; int j0 = 0, j1 = 0, j2 = 0;
    for (int i = 0; i < 768; ++i) {
      float v = sv[i]; int s = si_[i];
      if (v > w0)      { w2 = w1; j2 = j1; w1 = w0; j1 = j0; w0 = v; j0 = s; }
      else if (v > w1) { w2 = w1; j2 = j1; w1 = v; j1 = s; }
      else if (v > w2) { w2 = v; j2 = s; }
    }
    tv[b * 3] = w0; ti[b * 3] = j0;
    tv[b * 3 + 1] = w1; ti[b * 3 + 1] = j1;
    tv[b * 3 + 2] = w2; ti[b * 3 + 2] = j2;
  }
}

__global__ __launch_bounds__(128) void k_gather(const float* __restrict__ nc,
    const float* __restrict__ out_mem, const int* __restrict__ ti,
    __bf16* __restrict__ pair_bf, float* __restrict__ gathered) {
  int r = blockIdx.x, b = r / 3;
  int idx = ti[r];
  for (int c = threadIdx.x; c < D_DIM; c += 128) {
    float n = nc[(size_t)b * D_DIM + c];
    float g = out_mem[(size_t)idx * D_DIM + c];
    pair_bf[(size_t)r * 1024 + c] = (__bf16)n;
    pair_bf[(size_t)r * 1024 + 512 + c] = (__bf16)g;
    gathered[(size_t)r * D_DIM + c] = g;
  }
}

__global__ void k_pair_head(const float* __restrict__ ddh, const float* __restrict__ ddb1,
    const float* __restrict__ ddw2, const float* __restrict__ ddb2,
    const float* __restrict__ dsh, const float* __restrict__ dsb1,
    const float* __restrict__ dsw2, const float* __restrict__ dsb2,
    float* __restrict__ prob, float* __restrict__ strength) {
  int t = threadIdx.x;
  if (t >= 384) return;
  int which = t / 192, r = t % 192;
  const float* hp = which ? dsh : ddh;
  const float* b1 = which ? dsb1 : ddb1;
  const float* w2 = which ? dsw2 : ddw2;
  float acc = which ? dsb2[0] : ddb2[0];
  for (int j = 0; j < 256; ++j)
    acc += fmaxf(hp[(size_t)r * 256 + j] + b1[j], 0.f) * w2[j];
  (which ? strength : prob)[r] = sigmoidf(acc);
}

__global__ __launch_bounds__(128) void k_drift_mem(const float* __restrict__ nc,
    const float* __restrict__ gathered, const float* __restrict__ tv,
    const int* __restrict__ ti, const float* __restrict__ prob,
    const float* __restrict__ strength, float* __restrict__ out_mem) {
  int r = blockIdx.x, b = r / 3;
  float v = tv[r];
  if (!((v > 0.7f) && (v < 0.99f) && (prob[r] > 0.5f))) return;
  float st = strength[r];
  int idx = ti[r];
  for (int c = threadIdx.x; c < D_DIM; c += 128) {
    float g = gathered[(size_t)r * D_DIM + c];
    float n = nc[(size_t)b * D_DIM + c];
    out_mem[(size_t)idx * D_DIM + c] = (1.f - st) * g + st * (n + g) * 0.5f;
  }
}

__global__ void k_drift_new0(const float* __restrict__ nc, const float* __restrict__ gathered,
    const float* __restrict__ tv, const float* __restrict__ prob,
    const float* __restrict__ strength, float* __restrict__ drifted0) {
  int c = threadIdx.x;  // 512 threads
  float base = nc[c], dn = base;
  for (int k = 0; k < 3; ++k) {
    float v = tv[k];
    if ((v > 0.7f) && (v < 0.99f) && (prob[k] > 0.5f)) {
      float st = strength[k];
      float g = gathered[(size_t)k * D_DIM + c];
      dn = (1.f - st) * base + st * (base + g) * 0.5f;
    }
  }
  drifted0[c] = dn;
}

__global__ __launch_bounds__(256) void k_masked(const float* __restrict__ at,
    const float* __restrict__ es, const int* __restrict__ stepp, float* scal) {
  __shared__ unsigned long long rm[256], ra[256];
  int t = threadIdx.x, s = blockIdx.x * 256 + t;
  float step = (float)stepp[0];
  float at_s = at[s];
  float sa = step - at_s;
  bool recent = (at_s >= 0.f) && (sa < 3.f);
  float m = recent ? 0.f : es[s];
  unsigned sk = 0xFFFFFFFFu - (unsigned)s;
  rm[t] = ((unsigned long long)__float_as_uint(m) << 32) | sk;
  ra[t] = ((unsigned long long)__float_as_uint(sa) << 32) | sk;
  __syncthreads();
  for (int o = 128; o > 0; o >>= 1) {
    if (t < o) {
      rm[t] = rm[t] > rm[t + o] ? rm[t] : rm[t + o];
      ra[t] = ra[t] > ra[t + o] ? ra[t] : ra[t + o];
    }
    __syncthreads();
  }
  if (t == 0) {
    atomicMax((unsigned long long*)&scal[SC_MPACK], rm[0]);
    atomicMax((unsigned long long*)&scal[SC_APACK], ra[0]);
  }
}

__global__ void k_decide3(float* scal) {
  int* si = (int*)scal;
  unsigned long long mp = *(unsigned long long*)&scal[SC_MPACK];
  unsigned long long ap = *(unsigned long long*)&scal[SC_APACK];
  float mmax = __uint_as_float((unsigned)(mp >> 32));
  int useSlot = (mmax <= 0.f);
  int am = (int)(0xFFFFFFFFu - (unsigned)(mp & 0xFFFFFFFFull));
  int aa = (int)(0xFFFFFFFFu - (unsigned)(ap & 0xFFFFFFFFull));
  si[SC_WIDX] = si[SC_DOERASE] ? si[SC_VICTIM] : (useSlot ? aa : am);
  si[SC_USESLOT] = useSlot;
  si[SC_USEMASK] = si[SC_SHOULD] && !si[SC_DOERASE];
}

__global__ __launch_bounds__(256) void k_erase_out(const float* __restrict__ es,
    const float* __restrict__ at, const float* __restrict__ scal,
    const int* __restrict__ stepp, float* __restrict__ out_erase) {
  int s = blockIdx.x * 256 + threadIdx.x;
  if (s >= S_MEM) return;
  const int* si = (const int*)scal;
  float e = es[s];
  if (si[SC_USEMASK]) {
    float step = (float)stepp[0];
    float at_s = at[s];
    float sa = step - at_s;
    if (si[SC_USESLOT]) e = sa;
    else {
      bool recent = (at_s >= 0.f) && (sa < 3.f);
      e = recent ? 0.f : e;
    }
  }
  out_erase[s] = e;
}

__global__ void k_final(const float* __restrict__ drifted0, const float* __restrict__ scal,
                        const int* __restrict__ stepp, float* out_mem, float* out_at) {
  const int* si = (const int*)scal;
  if (!si[SC_SHOULD]) return;
  int w = si[SC_WIDX];
  out_mem[(size_t)w * D_DIM + threadIdx.x] = drifted0[threadIdx.x];
  if (threadIdx.x == 0) out_at[w] = (float)stepp[0];
}

__global__ __launch_bounds__(256) void k_changes(const float* __restrict__ a,
                                                 const float* __restrict__ b, float* scal) {
  __shared__ float red[256];
  int t = threadIdx.x;
  float s = 0.f;
  const float4* A = (const float4*)a;
  const float4* B = (const float4*)b;
  const size_t n4 = (size_t)S_MEM * D_DIM / 4;
  for (size_t i = (size_t)blockIdx.x * 256 + t; i < n4; i += (size_t)gridDim.x * 256) {
    float4 x = A[i], y = B[i];
    s += fabsf(x.x - y.x) + fabsf(x.y - y.y) + fabsf(x.z - y.z) + fabsf(x.w - y.w);
  }
  red[t] = s; __syncthreads();
  for (int o = 128; o > 0; o >>= 1) { if (t < o) red[t] += red[t + o]; __syncthreads(); }
  if (t == 0 && red[0] != 0.f) atomicAdd(&scal[SC_CHSUM], red[0]);
}

__global__ void k_fin(const float* scal, float* out_rc) {
  out_rc[0] = scal[SC_CHSUM] / (float)((size_t)S_MEM * D_DIM);
}

extern "C" void kernel_launch(void* const* d_in, const int* in_sizes, int n_in,
                              void* d_out, int out_size, void* d_ws, size_t ws_size,
                              hipStream_t stream) {
  (void)in_sizes; (void)n_in; (void)out_size; (void)ws_size;
  const float* nc   = (const float*)d_in[0];
  const float* qry  = (const float*)d_in[1];
  const float* mem  = (const float*)d_in[2];
  const float* at_i = (const float*)d_in[3];
  const int*   stp  = (const int*)d_in[4];
  const float* srw1 = (const float*)d_in[5];
  const float* srb1 = (const float*)d_in[6];
  const float* srg  = (const float*)d_in[7];
  const float* srbe = (const float*)d_in[8];
  const float* srw2 = (const float*)d_in[9];
  const float* srb2 = (const float*)d_in[10];
  const float* snw  = (const float*)d_in[11];
  const float* snb  = (const float*)d_in[12];
  const float* sgw  = (const float*)d_in[13];
  const float* sgb  = (const float*)d_in[14];
  const float* elw  = (const float*)d_in[15];
  const float* elb  = (const float*)d_in[16];
  const float* ecw  = (const float*)d_in[17];
  const float* ecb  = (const float*)d_in[18];
  const float* egw  = (const float*)d_in[19];
  const float* egb  = (const float*)d_in[20];
  const float* ddw1 = (const float*)d_in[21];
  const float* ddb1 = (const float*)d_in[22];
  const float* ddw2 = (const float*)d_in[23];
  const float* ddb2 = (const float*)d_in[24];
  const float* dsw1 = (const float*)d_in[25];
  const float* dsb1 = (const float*)d_in[26];
  const float* dsw2 = (const float*)d_in[27];
  const float* dsb2 = (const float*)d_in[28];

  float* out = (float*)d_out;
  float* out_mem   = out;
  float* out_at    = out + (size_t)S_MEM * D_DIM;
  float* out_erase = out_at + S_MEM;
  float* out_score = out_erase + S_MEM;
  float* out_nov   = out_score + B_DIM;
  float* out_rc    = out_nov + B_DIM;

  // workspace bump allocator
  char* p = (char*)d_ws;
  auto alloc = [&](size_t bytes) -> char* {
    char* r = p; p += (bytes + 255) & ~(size_t)255; return r;
  };
  float*  scal      = (float*)alloc(4096);
  __bf16* nn_bf     = (__bf16*)alloc((size_t)B_DIM * D_DIM * 2);
  __bf16* nc_bf     = (__bf16*)alloc((size_t)B_DIM * D_DIM * 2);
  __bf16* comb_bf   = (__bf16*)alloc((size_t)B_DIM * 1024 * 2);
  __bf16* srw1_bf   = (__bf16*)alloc((size_t)1024 * 256 * 2);
  __bf16* srw2_bf   = (__bf16*)alloc((size_t)256 * 128 * 2);
  __bf16* snw_bf    = (__bf16*)alloc((size_t)512 * 128 * 2);
  __bf16* ecw_bf    = (__bf16*)alloc((size_t)512 * 32 * 2);
  __bf16* ddw1_bf   = (__bf16*)alloc((size_t)1024 * 256 * 2);
  __bf16* dsw1_bf   = (__bf16*)alloc((size_t)1024 * 256 * 2);
  float*  Hpre      = (float*)alloc((size_t)B_DIM * 256 * 4);
  __bf16* hrelu_bf  = (__bf16*)alloc((size_t)B_DIM * 256 * 2);
  float*  relpre    = (float*)alloc((size_t)B_DIM * 128 * 4);
  float*  nfpre     = (float*)alloc((size_t)B_DIM * 128 * 4);
  float*  sims      = (float*)alloc((size_t)B_DIM * S_MEM * 4);
  float*  norms     = (float*)alloc((size_t)S_MEM * 4);
  float*  conf32    = (float*)alloc((size_t)S_MEM * 32 * 4);
  float*  es        = (float*)alloc((size_t)S_MEM * 4);
  float*  tv        = (float*)alloc((size_t)B_DIM * 3 * 4);
  int*    ti        = (int*)alloc((size_t)B_DIM * 3 * 4);
  __bf16* pair_bf   = (__bf16*)alloc((size_t)192 * 1024 * 2);
  float*  gathered  = (float*)alloc((size_t)192 * D_DIM * 4);
  float*  ddh       = (float*)alloc((size_t)192 * 256 * 4);
  float*  dsh       = (float*)alloc((size_t)192 * 256 * 4);
  float*  prob      = (float*)alloc((size_t)192 * 4);
  float*  strength  = (float*)alloc((size_t)192 * 4);
  float*  drifted0  = (float*)alloc((size_t)D_DIM * 4);

  // seed outputs with copies of the bank / access times
  (void)hipMemcpyAsync(out_mem, mem, (size_t)S_MEM * D_DIM * 4, hipMemcpyDeviceToDevice, stream);
  (void)hipMemcpyAsync(out_at, at_i, (size_t)S_MEM * 4, hipMemcpyDeviceToDevice, stream);

  k_init<<<1, 64, 0, stream>>>(scal);

  k_cvt<<<256, 256, 0, stream>>>(srw1, srw1_bf, 1024 * 256);
  k_cvt<<<64, 256, 0, stream>>>(srw2, srw2_bf, 256 * 128);
  k_cvt<<<128, 256, 0, stream>>>(snw, snw_bf, 512 * 128);
  k_cvt<<<32, 256, 0, stream>>>(ecw, ecw_bf, 512 * 32);
  k_cvt<<<256, 256, 0, stream>>>(ddw1, ddw1_bf, 1024 * 256);
  k_cvt<<<256, 256, 0, stream>>>(dsw1, dsw1_bf, 1024 * 256);

  k_prep<<<B_DIM, 128, 0, stream>>>(nc, qry, nn_bf, nc_bf, comb_bf);

  // fused heavy pass: norms + sims (B x S) + conf (S x 32), bank read once
  k_heavy<<<S_MEM / 16, 128, 0, stream>>>(mem, nn_bf, ecw_bf, ecb, sims, norms, conf32, scal);

  // store-gate MLP chain (bf16 WMMA)
  k_wmma_gemm<<<16, 128, 0, stream>>>(comb_bf, srw1_bf, Hpre, 64, 256, 1024);
  k_b_ln<<<B_DIM, 256, 0, stream>>>(Hpre, srb1, srg, srbe, hrelu_bf);
  k_wmma_gemm<<<8, 128, 0, stream>>>(hrelu_bf, srw2_bf, relpre, 64, 128, 256);
  k_wmma_gemm<<<8, 128, 0, stream>>>(nc_bf, snw_bf, nfpre, 64, 128, 512);
  k_b_score<<<1, 64, 0, stream>>>(relpre, srb2, nfpre, snb, sgw, sgb, out_score, scal);

  // global reductions + decisions
  k_age<<<S_MEM / 256, 256, 0, stream>>>(at_i, stp, scal);
  k_victim<<<S_MEM / 256, 256, 0, stream>>>(at_i, norms, stp, scal);
  k_novelty<<<B_DIM, 256, 0, stream>>>(sims, out_nov, scal);
  k_decide1<<<1, 1, 0, stream>>>(scal);
  k_cond<<<2048, 256, 0, stream>>>(sims, norms, scal);
  k_decide2<<<1, 1, 0, stream>>>(scal);

  // erase + erase scores
  k_erase_apply<<<1, 512, 0, stream>>>(out_mem, out_at, scal);
  k_es<<<S_MEM / 256, 256, 0, stream>>>(out_at, conf32, elw, elb, egw, egb, ecb, scal, stp, es);

  // consolidation: top-3, pair MLPs (bf16 WMMA), drift
  k_topk<<<B_DIM, 256, 0, stream>>>(sims, scal, ti, tv);
  k_gather<<<192, 128, 0, stream>>>(nc, out_mem, ti, pair_bf, gathered);
  k_wmma_gemm<<<48, 128, 0, stream>>>(pair_bf, ddw1_bf, ddh, 192, 256, 1024);
  k_wmma_gemm<<<48, 128, 0, stream>>>(pair_bf, dsw1_bf, dsh, 192, 256, 1024);
  k_pair_head<<<1, 384, 0, stream>>>(ddh, ddb1, ddw2, ddb2, dsh, dsb1, dsw2, dsb2, prob, strength);
  k_drift_mem<<<192, 128, 0, stream>>>(nc, gathered, tv, ti, prob, strength, out_mem);
  k_drift_new0<<<1, 512, 0, stream>>>(nc, gathered, tv, prob, strength, drifted0);

  // write-slot selection, erase_out, final store
  k_masked<<<S_MEM / 256, 256, 0, stream>>>(out_at, es, stp, scal);
  k_decide3<<<1, 1, 0, stream>>>(scal);
  k_erase_out<<<S_MEM / 256, 256, 0, stream>>>(es, out_at, scal, stp, out_erase);
  k_final<<<1, 512, 0, stream>>>(drifted0, scal, stp, out_mem, out_at);

  // recent_changes
  k_changes<<<2048, 256, 0, stream>>>(out_mem, mem, scal);
  k_fin<<<1, 1, 0, stream>>>(scal, out_rc);
}
